// ConvTransposeNd_groups_4518305595870
// MI455X (gfx1250) — matrix operand order repaired and checked by reading, not compiled, not run on hardware
//
#include <hip/hip_runtime.h>

typedef __attribute__((ext_vector_type(16))) _Float16 v16h;
typedef __attribute__((ext_vector_type(8)))  _Float16 v8h;
typedef __attribute__((ext_vector_type(8)))  float    v8f;

// Sizes (fixed by the reference problem)
#define NB    4
#define CIN   64
#define COUT  64
#define LL    12          // input spatial extent per dim
#define OO    25          // output spatial extent per dim = (12-1)*2+3
#define KVOL  81          // 3^4
#define XT_HALFS (NB*CIN*LL*LL*LL*LL)   // 5,308,416
#define WT_HALFS (KVOL*COUT*CIN)        // 331,776

// LDS slab geometry: rows indexed by (l4+1) in [0,16] -> 17 rows of 64 halfs.
#define SLAB_ROWS   17
#define SLAB_HALFS  (SLAB_ROWS*64)      // 1088 halfs = 2176 B (16B aligned)
#define SLAB_DW     (SLAB_HALFS/2)      // 544 dwords

// ---------------------------------------------------------------------------
// Pre-pass 1: weights fp32 [Cin][Cout][3][3][3][3] -> f16 wt[kidx][o][i]
// ---------------------------------------------------------------------------
__global__ __launch_bounds__(256) void prep_w(const float* __restrict__ W,
                                              _Float16* __restrict__ wt) {
  int ot = blockIdx.x * 256 + threadIdx.x;
  if (ot >= WT_HALFS) return;
  int i    = ot & 63;
  int o    = (ot >> 6) & 63;
  int kidx = ot >> 12;
  wt[ot] = (_Float16)W[(i * 64 + o) * KVOL + kidx];
}

// ---------------------------------------------------------------------------
// Pre-pass 2: x fp32 [B][Cin][12][12][12][12] -> f16 xt[b][l1][l2][l3][l4][i]
// (channels INNERMOST: per (b,l1,l2,l3) a contiguous 12x64-half slab whose
//  rows are l4 and columns are Cin -> B operands become contiguous 32B runs)
// ---------------------------------------------------------------------------
__global__ __launch_bounds__(256) void prep_x(const float* __restrict__ X,
                                              _Float16* __restrict__ xt) {
  int d = blockIdx.x * 256 + threadIdx.x;
  if (d >= XT_HALFS) return;
  int i    = d & 63;
  int t    = d >> 6;
  int l4   = t % 12;
  int rest = t / 12;                 // b*1728 + l1*144 + l2*12 + l3
  int l3   = rest % 12;
  int l2   = (rest / 12) % 12;
  int l1   = (rest / 144) % 12;
  int b    = rest / 1728;
  xt[d] = (_Float16)X[((b * 64 + i) * 20736) + l1 * 1728 + l2 * 144 + l3 * 12 + l4];
}

// ---------------------------------------------------------------------------
// Main kernel: one block (128 thr = 4 waves) per (b, y1, y2, y3, p4).
//   M = Cout = 64 (4 waves x 16 rows), N = 16 (m4), K = 64 * taps.
// ---------------------------------------------------------------------------
__global__ __launch_bounds__(128) void convt4d_wmma(
    const _Float16* __restrict__ xt, const _Float16* __restrict__ wt,
    const float* __restrict__ bias, float* __restrict__ out) {
  __shared__ __align__(16) _Float16 lds[8 * SLAB_HALFS];   // 17408 bytes
  unsigned* ldsU = (unsigned*)lds;

  const int tid  = threadIdx.x;
  const int lane = tid & 31;
  const int wv   = tid >> 5;                      // wave id 0..3 -> Cout tile
  const int n    = lane & 15;                     // N col / A row (M) in tile
  const int half = lane >> 4;                     // lane group 0/1

  int xg = blockIdx.x;
  const int y3 = xg % 25; xg /= 25;
  const int y2 = xg % 25; xg /= 25;
  const int y1 = xg % 25;
  const int b  = xg / 25;
  const int p4 = blockIdx.y;
  const int p1 = y1 & 1, p2 = y2 & 1, p3 = y3 & 1;
  const int m1 = y1 >> 1, m2 = y2 >> 1, m3 = y3 >> 1;

  // zero LDS (8*544 dwords)
  for (int t = tid; t < 8 * SLAB_DW; t += 128) ldsU[t] = 0u;
  __syncthreads();

  // stage up to 8 input slabs (j1,j2,j3) from xt into LDS rows 1..12
  // straight 16-byte copies: src and dst are both [l4][i] with i innermost.
  for (int s = 0; s < 8; ++s) {
    int j1 = (s >> 2) & 1, j2 = (s >> 1) & 1, j3 = s & 1;
    if ((p1 & j1) | (p2 & j2) | (p3 & j3)) continue;      // tap absent (odd parity)
    int l1 = m1 - j1, l2 = m2 - j2, l3 = m3 - j3;
    if ((unsigned)l1 >= 12u || (unsigned)l2 >= 12u || (unsigned)l3 >= 12u) continue;
    int slab_src = ((b * 12 + l1) * 12 + l2) * 12 + l3;
    const uint4* src = (const uint4*)xt + slab_src * 96;  // 12*64 halfs = 96*16B
    uint4* dst = (uint4*)(ldsU + s * SLAB_DW + 32);       // row 1 (=l4 0) start
    for (int t = tid; t < 96; t += 128) dst[t] = src[t];
  }
  __syncthreads();

  v8f c = {};
  const int t1 = p1 ? 1 : 2, t2 = p2 ? 1 : 2, t3 = p3 ? 1 : 2, t4 = p4 ? 1 : 2;

  for (int j1 = 0; j1 < t1; ++j1)
    for (int j2 = 0; j2 < t2; ++j2)
      for (int j3 = 0; j3 < t3; ++j3) {
        int l1 = m1 - j1, l2 = m2 - j2, l3 = m3 - j3;
        if ((unsigned)l1 >= 12u || (unsigned)l2 >= 12u || (unsigned)l3 >= 12u) continue;
        const _Float16* slab = &lds[(j1 * 4 + j2 * 2 + j3) * SLAB_HALFS];
        for (int j4 = 0; j4 < t4; ++j4) {
          const int k1 = p1 + 2 * j1, k2 = p2 + 2 * j2, k3 = p3 + 2 * j3, k4 = p4 + 2 * j4;
          const int kidx = ((k1 * 3 + k2) * 3 + k3) * 3 + k4;
          // A row for this lane: wt[kidx][o = wv*16 + n][0..63]
          const _Float16* Arow = wt + kidx * 4096 + (wv * 16 + n) * 64;
          // B row base for this lane: slab[(l4+1)*64 + ...], l4 = n - j4
          const _Float16* Brow = slab + ((n - j4) + 1) * 64 + half * 16;
          __builtin_prefetch(Arow + 4096, 0, 1);   // next tap's weight tile
          #pragma unroll
          for (int kk = 0; kk < 2; ++kk) {         // K = 64 -> 2 chunks of 32
            // A (16x32 f16): halves 0-15 = K half*16+0..15 (contiguous)
            v8h a0 = *(const v8h*)(Arow + kk * 32 + half * 16);
            v8h a1 = *(const v8h*)(Arow + kk * 32 + half * 16 + 8);
            v16h a = __builtin_shufflevector(a0, a1, 0, 1, 2, 3, 4, 5, 6, 7,
                                                     8, 9, 10, 11, 12, 13, 14, 15);
            // B (32x16 f16): lane's 16 K values are 32 contiguous bytes
            v16h bm = *(const v16h*)(Brow + kk * 32);
            c = __builtin_amdgcn_wmma_f32_16x16x32_f16(
                    false, a, false, bm, (short)0, c, false, false);
          }
        }
      }

  // store: C/D layout -> VGPR r holds M = r + half*8; N = lane%16 = m4
  const int extent = p4 ? 12 : 13;
  if (n < extent) {
    const int y4 = 2 * n + p4;
    #pragma unroll
    for (int r = 0; r < 8; ++r) {
      int o = wv * 16 + half * 8 + r;
      int idx = (b * 64 + o) * 390625 + y1 * 15625 + y2 * 625 + y3 * 25 + y4;
      out[idx] = c[r] + bias[o];
    }
  }
}

// ---------------------------------------------------------------------------
extern "C" void kernel_launch(void* const* d_in, const int* in_sizes, int n_in,
                              void* d_out, int out_size, void* d_ws, size_t ws_size,
                              hipStream_t stream) {
  const float* x    = (const float*)d_in[0];
  const float* wgt  = (const float*)d_in[1];
  const float* bias = (const float*)d_in[2];
  float* out = (float*)d_out;

  _Float16* xt = (_Float16*)d_ws;            // 10,616,832 B
  _Float16* wt = xt + XT_HALFS;              // +663,552 B  (~11.3 MB total)

  prep_x<<<XT_HALFS / 256, 256, 0, stream>>>(x, xt);
  prep_w<<<WT_HALFS / 256, 256, 0, stream>>>(wgt, wt);

  dim3 grid(NB * 25 * 25 * 25, 2);           // (b,y1,y2,y3) x parity(y4)
  convt4d_wmma<<<grid, 128, 0, stream>>>(xt, wt, bias, out);
}
